// QNN_64115271795096
// MI455X (gfx1250) — matrix operand order repaired and checked by reading, not compiled, not run on hardware
//
#include <hip/hip_runtime.h>
#include <hip/hip_bf16.h>

#define TT   128
#define BB   64
#define DD   64
#define CCD  10
#define INDIM 300

typedef __attribute__((ext_vector_type(16))) _Float16 v16h;
typedef __attribute__((ext_vector_type(8)))  _Float16 v8h;
typedef __attribute__((ext_vector_type(8)))  float    v8f;

// ---------------------------------------------------------------------------
// WMMA fragment helpers (layouts per CDNA5 ISA 7.12.2, wave32)
// A operand: 16x32 f16, lane&15 = M row; lanes<16 hold K 0-7 & 16-23,
//            lanes>=16 hold K 8-15 & 24-31 -> two contiguous 8-half runs.
// B operand expects B^T row-major ([N][K]); lane&15 = N, same K split as A:
//            lanes<16 K 0-7 & 16-23, lanes>=16 K 8-15 & 24-31.
// C/D f32:   element v -> (M = v + 8*(lane>=16), N = lane&15).
// ---------------------------------------------------------------------------
__device__ __forceinline__ v16h load_a_frag(const _Float16* M, int mBase, int kBase, int lane) {
  const int row = mBase + (lane & 15);
  const int k0  = kBase + ((lane >> 4) << 3);        // 0 or 8
  const v8h lo = *(const v8h*)(M + row * DD + k0);
  const v8h hi = *(const v8h*)(M + row * DD + k0 + 16);
  v16h r;
#pragma unroll
  for (int i = 0; i < 8; ++i) { r[i] = lo[i]; r[i + 8] = hi[i]; }
  return r;
}

__device__ __forceinline__ v16h load_b_frag(const _Float16* S, int nBase, int kBase, int lane) {
  const int row = nBase + (lane & 15);
  const int k0  = kBase + ((lane >> 4) << 3);        // 0 or 8
  const v8h lo = *(const v8h*)(S + row * DD + k0);
  const v8h hi = *(const v8h*)(S + row * DD + k0 + 16);
  v16h r;
#pragma unroll
  for (int i = 0; i < 8; ++i) { r[i] = lo[i]; r[i + 8] = hi[i]; }
  return r;
}

// D(16x16) += A * S^T  where S is row-major [N][K] (i.e. B = S^T).
// f16 WMMA has no A/B negation modifiers on gfx1250 (NEG = CNeg,0,0),
// so all sign handling is done algebraically / at stores.
__device__ __forceinline__ v8f tile_mm(const _Float16* A, const _Float16* S,
                                       int m, int n, v8f c, int lane) {
  v16h a0 = load_a_frag(A, m, 0, lane);
  v16h b0 = load_b_frag(S, n, 0, lane);
  c = __builtin_amdgcn_wmma_f32_16x16x32_f16(false, a0, false, b0, (short)0, c, false, false);
  v16h a1 = load_a_frag(A, m, 32, lane);
  v16h b1 = load_b_frag(S, n, 32, lane);
  c = __builtin_amdgcn_wmma_f32_16x16x32_f16(false, a1, false, b1, (short)0, c, false, false);
  return c;
}

__device__ __forceinline__ void store_c_h(_Float16* D, int m, int n, v8f c, int lane, float scale) {
  const int col = n + (lane & 15);
  const int rb  = m + ((lane >> 4) << 3);
#pragma unroll
  for (int v = 0; v < 8; ++v) D[(rb + v) * DD + col] = (_Float16)(c[v] * scale);
}

// ---------------------------------------------------------------------------
// 1) Front-end: projections + L2-norm + cos/sin + rho = |s><s|  -> f16 seq
//    one 64-thread block per (b,t)
// ---------------------------------------------------------------------------
__global__ __launch_bounds__(64) void qrnn_front(
    const float* __restrict__ x, const float* __restrict__ amp_w, const float* __restrict__ amp_b,
    const float* __restrict__ phase_w, const float* __restrict__ phase_b,
    _Float16* __restrict__ seqR, _Float16* __restrict__ seqI)
{
  __shared__ float s_a[DD], s_r[DD], s_i[DD];
  const int blk = blockIdx.x;
  const int b = blk / TT, t = blk % TT;
  const int d = threadIdx.x;
  const float* xr = x + ((size_t)b * TT + t) * INDIM;
  float aa = amp_b[d], pp = phase_b[d];
  for (int k = 0; k < INDIM; ++k) {
    const float xv = xr[k];
    aa += xv * amp_w[k * DD + d];
    pp += xv * phase_w[k * DD + d];
  }
  s_a[d] = aa * aa;
  __syncthreads();
  for (int off = 32; off > 0; off >>= 1) {
    if (d < off) s_a[d] += s_a[d + off];
    __syncthreads();
  }
  const float an = aa * rsqrtf(s_a[0]);
  s_r[d] = an * cosf(pp);
  s_i[d] = an * sinf(pp);
  __syncthreads();
  const size_t base = ((size_t)t * BB + b) * (DD * DD);
  const float srd = s_r[d], sid = s_i[d];
  for (int e = 0; e < DD; ++e) {
    const float sre = s_r[e], sie = s_i[e];
    seqR[base + d * DD + e] = (_Float16)(srd * sre + sid * sie);   // symmetric
    seqI[base + d * DD + e] = (_Float16)(sid * sre - srd * sie);   // antisymmetric
  }
}

// ---------------------------------------------------------------------------
// 2) Ux rho Ux^T for all (t,b) tiles in parallel (WMMA).
//    rho Hermitian: rr^T = rr, ri^T = -ri.  Feeding ri row-major as B gives
//    -Ux@ri; fix the sign at the f16 store.
// ---------------------------------------------------------------------------
__global__ __launch_bounds__(256) void qrnn_uconjx(
    const float* __restrict__ Ux_f32,
    const _Float16* __restrict__ inR, const _Float16* __restrict__ inI,
    _Float16* __restrict__ outR, _Float16* __restrict__ outI)
{
  __shared__ alignas(16) _Float16 sU [DD * DD];
  __shared__ alignas(16) _Float16 sRr[DD * DD];
  __shared__ alignas(16) _Float16 sRi[DD * DD];
  __shared__ alignas(16) _Float16 sT1[DD * DD];
  __shared__ alignas(16) _Float16 sT2[DD * DD];
  const size_t base = (size_t)blockIdx.x * (DD * DD);
  const int tid = threadIdx.x, wave = tid >> 5, lane = tid & 31;

  for (int i = tid; i < DD * DD; i += 256) sU[i] = (_Float16)Ux_f32[i];
  const uint4* gr = (const uint4*)(inR + base);
  const uint4* gi = (const uint4*)(inI + base);
  for (int i = tid; i < (DD * DD) / 8; i += 256) {
    ((uint4*)sRr)[i] = gr[i];
    ((uint4*)sRi)[i] = gi[i];
  }
  __syncthreads();
  // t1 = Ux @ rr  (rr^T = rr),  t2 = Ux @ ri  (B=ri gives -Ux@ri -> store -1x)
#pragma unroll
  for (int q = 0; q < 2; ++q) {
    const int ti = wave + q * 8;
    const int m = (ti >> 2) * 16, n = (ti & 3) * 16;
    v8f c = {}; c = tile_mm(sU, sRr, m, n, c, lane); store_c_h(sT1, m, n, c, lane,  1.0f);
    v8f d = {}; d = tile_mm(sU, sRi, m, n, d, lane); store_c_h(sT2, m, n, d, lane, -1.0f);
  }
  __syncthreads();
  // ux = t @ Ux^T : B-layout of Ux^T is Ux row-major
#pragma unroll
  for (int q = 0; q < 2; ++q) {
    const int ti = wave + q * 8;
    const int m = (ti >> 2) * 16, n = (ti & 3) * 16;
    v8f c = {}; c = tile_mm(sT1, sU, m, n, c, lane);
    v8f d = {}; d = tile_mm(sT2, sU, m, n, d, lane);
    const int col = n + (lane & 15);
    const int rb  = m + ((lane >> 4) << 3);
#pragma unroll
    for (int v = 0; v < 8; ++v) {
      const int idx = (rb + v) * DD + col;
      outR[base + idx] = (_Float16)c[v];
      outI[base + idx] = (_Float16)d[v];
    }
  }
}

// ---------------------------------------------------------------------------
// 3) Sequential QRNN scan: one block per batch element, h resident in LDS.
//    Per step: uh = Uh h Uh^T, mix with precomputed ux, then rho^2 via the
//    Hermitian trick:  r2 = nr@nr - ni@ni  (B-operand ni natively = -ni),
//                      i2 = X - X^T with X = ni@nr  (since nr@ni = -X^T).
// ---------------------------------------------------------------------------
__global__ __launch_bounds__(256) void qrnn_scan(
    const float* __restrict__ Uh_f32, const float* __restrict__ lam_p,
    const _Float16* __restrict__ uxR, const _Float16* __restrict__ uxI,
    _Float16* __restrict__ seqR, _Float16* __restrict__ seqI)
{
  __shared__ alignas(16) _Float16 sUh[DD * DD];
  __shared__ alignas(16) _Float16 sHr[DD * DD];
  __shared__ alignas(16) _Float16 sHi[DD * DD];
  __shared__ alignas(16) _Float16 sT1[DD * DD];
  __shared__ alignas(16) _Float16 sT2[DD * DD];
  __shared__ alignas(16) _Float16 sNr[DD * DD];
  __shared__ alignas(16) _Float16 sNi[DD * DD];
  __shared__ float trsum;

  const int b = blockIdx.x;
  const int tid = threadIdx.x, wave = tid >> 5, lane = tid & 31;
  const float lam = 1.0f / (1.0f + expf(-lam_p[0]));

  for (int i = tid; i < DD * DD; i += 256) {
    sUh[i] = (_Float16)Uh_f32[i];
    sHr[i] = (_Float16)(((i / DD) == (i % DD)) ? (1.0f / DD) : 0.0f);
    sHi[i] = (_Float16)0.0f;
  }
  __syncthreads();

  for (int t = 0; t < TT; ++t) {
    const size_t base = ((size_t)t * BB + b) * (DD * DD);
    if (t + 1 < TT) {
      __builtin_prefetch(uxR + base + (size_t)BB * DD * DD, 0, 0);
      __builtin_prefetch(uxI + base + (size_t)BB * DD * DD, 0, 0);
    }
    // stage 1: t1 = Uh @ hr ; t2 = Uh @ hi  (B=hi gives -Uh@hi -> store -1x)
#pragma unroll
    for (int q = 0; q < 2; ++q) {
      const int ti = wave + q * 8;
      const int m = (ti >> 2) * 16, n = (ti & 3) * 16;
      v8f c = {}; c = tile_mm(sUh, sHr, m, n, c, lane); store_c_h(sT1, m, n, c, lane,  1.0f);
      v8f d = {}; d = tile_mm(sUh, sHi, m, n, d, lane); store_c_h(sT2, m, n, d, lane, -1.0f);
    }
    __syncthreads();
    // stage 2: uh = t @ Uh^T, then n = lam*ux + (1-lam)*uh  -> sNr/sNi f16
#pragma unroll
    for (int q = 0; q < 2; ++q) {
      const int ti = wave + q * 8;
      const int m = (ti >> 2) * 16, n = (ti & 3) * 16;
      v8f c = {}; c = tile_mm(sT1, sUh, m, n, c, lane);
      v8f d = {}; d = tile_mm(sT2, sUh, m, n, d, lane);
      const int col = n + (lane & 15);
      const int rb  = m + ((lane >> 4) << 3);
#pragma unroll
      for (int v = 0; v < 8; ++v) {
        const int idx = (rb + v) * DD + col;
        sNr[idx] = (_Float16)(lam * (float)uxR[base + idx] + (1.0f - lam) * c[v]);
        sNi[idx] = (_Float16)(lam * (float)uxI[base + idx] + (1.0f - lam) * d[v]);
      }
    }
    if (tid == 0) trsum = 0.0f;
    __syncthreads();
    // stage 3: r2 (registers) and X = ni@nr (f16, into sT1; sT1 is dead here)
    v8f cc[2];
    int mm[2], nn[2];
#pragma unroll
    for (int q = 0; q < 2; ++q) {
      const int ti = wave + q * 8;
      const int m = (ti >> 2) * 16, n = (ti & 3) * 16;
      mm[q] = m; nn[q] = n;
      v8f c = {};
      c = tile_mm(sNr, sNr, m, n, c, lane);     // + nr @ nr   (nr^T = nr)
      c = tile_mm(sNi, sNi, m, n, c, lane);     // - ni @ ni   (ni^T = -ni)
      cc[q] = c;
      v8f xx = {};
      xx = tile_mm(sNi, sNr, m, n, xx, lane);   // X = ni @ nr
      store_c_h(sT1, m, n, xx, lane, 1.0f);
      if (m == n) {                             // diagonal tile -> trace(r2)
#pragma unroll
        for (int v = 0; v < 8; ++v) {
          if ((lane < 16 && lane == v) || (lane >= 16 && (lane - 24) == v))
            atomicAdd(&trsum, c[v]);
        }
      }
    }
    __syncthreads();
    const float invtr = 1.0f / trsum;
    // stage 4: h' = 0.8*r2/tr + 0.2*nr ;  i2 = X - X^T
#pragma unroll
    for (int q = 0; q < 2; ++q) {
      const int col = nn[q] + (lane & 15);
      const int rb  = mm[q] + ((lane >> 4) << 3);
#pragma unroll
      for (int v = 0; v < 8; ++v) {
        const int row = rb + v;
        const int idx  = row * DD + col;
        const int idxT = col * DD + row;
        const float i2v = (float)sT1[idx] - (float)sT1[idxT];
        const _Float16 hr16 = (_Float16)(0.8f * cc[q][v] * invtr + 0.2f * (float)sNr[idx]);
        const _Float16 hi16 = (_Float16)(0.8f * i2v      * invtr + 0.2f * (float)sNi[idx]);
        sHr[idx] = hr16; sHi[idx] = hi16;
        seqR[base + idx] = hr16; seqI[base + idx] = hi16;
      }
    }
    __syncthreads();
  }
}

// ---------------------------------------------------------------------------
// 4) Tail: QDense (W^T rho W), Lambda-mix, trace-norm, measurement, log.
// ---------------------------------------------------------------------------
__global__ __launch_bounds__(128) void qrnn_tail(
    const _Float16* __restrict__ seqR, const _Float16* __restrict__ seqI,
    const float* __restrict__ dense_w, const float* __restrict__ dlam_p,
    const float* __restrict__ meas, float* __restrict__ out)
{
  __shared__ float M1r[DD * CCD], M1i[DD * CCD];
  __shared__ float sgr[CCD * CCD], sgi[CCD * CCD];
  __shared__ float invtr_s;
  const int blk = blockIdx.x;
  const int t = blk / BB, b = blk % BB;
  const int tid = threadIdx.x;
  const size_t base = ((size_t)t * BB + b) * (DD * DD);
  const float dlam = 1.0f / (1.0f + expf(-dlam_p[0]));

  for (int e = tid; e < DD * CCD; e += 128) {        // M1 = rho @ W
    const int d = e / CCD, c = e % CCD;
    float ar = 0.0f, ai = 0.0f;
    for (int k = 0; k < DD; ++k) {
      const float w = dense_w[k * CCD + c];
      ar += (float)seqR[base + d * DD + k] * w;
      ai += (float)seqI[base + d * DD + k] * w;
    }
    M1r[e] = ar; M1i[e] = ai;
  }
  __syncthreads();
  for (int e = tid; e < CCD * CCD; e += 128) {       // sigma = W^T @ M1, mixed
    const int c = e / CCD, f = e % CCD;
    float ar = 0.0f, ai = 0.0f;
    for (int d = 0; d < DD; ++d) {
      const float w = dense_w[d * CCD + c];
      ar += w * M1r[d * CCD + f];
      ai += w * M1i[d * CCD + f];
    }
    sgr[e] = dlam * ar + (1.0f - dlam) * ((c == f) ? (1.0f / CCD) : 0.0f);
    sgi[e] = dlam * ai;
  }
  __syncthreads();
  if (tid == 0) {
    float tr = 0.0f;
    for (int c = 0; c < CCD; ++c) tr += sgr[c * CCD + c];
    invtr_s = 1.0f / tr;
  }
  __syncthreads();
  if (tid < CCD) {
    const int k = tid;
    float vr[CCD], vi[CCD], nrm = 0.0f;
    for (int c = 0; c < CCD; ++c) {
      vr[c] = meas[(k * CCD + c) * 2 + 0];
      vi[c] = meas[(k * CCD + c) * 2 + 1];
      nrm += vr[c] * vr[c] + vi[c] * vi[c];
    }
    const float inv2 = 1.0f / nrm;                   // |v|^-2 applied to p
    float p = 0.0f;
    for (int c = 0; c < CCD; ++c)
      for (int dI = 0; dI < CCD; ++dI)
        p += (vr[c] * vr[dI] + vi[c] * vi[dI]) * sgr[c * CCD + dI]
           + (vi[c] * vr[dI] - vr[c] * vi[dI]) * sgi[c * CCD + dI];
    out[((size_t)b * TT + t) * CCD + k] = logf(p * inv2 * invtr_s);
  }
}

// ---------------------------------------------------------------------------
extern "C" void kernel_launch(void* const* d_in, const int* in_sizes, int n_in,
                              void* d_out, int out_size, void* d_ws, size_t ws_size,
                              hipStream_t stream) {
  (void)in_sizes; (void)n_in; (void)out_size; (void)ws_size;
  const float* x       = (const float*)d_in[0];
  const float* amp_w   = (const float*)d_in[1];
  const float* amp_b   = (const float*)d_in[2];
  const float* phase_w = (const float*)d_in[3];
  const float* phase_b = (const float*)d_in[4];
  const float* Ux      = (const float*)d_in[5];
  const float* Uh      = (const float*)d_in[6];
  const float* lam_p   = (const float*)d_in[7];
  const float* dense_w = (const float*)d_in[8];
  const float* dlam_p  = (const float*)d_in[9];
  const float* meas    = (const float*)d_in[10];

  const size_t NSEQ = (size_t)TT * BB * DD * DD;     // 33.5M halves per buffer
  _Float16* seqR = (_Float16*)d_ws;
  _Float16* seqI = seqR + NSEQ;
  _Float16* uxR  = seqI + NSEQ;
  _Float16* uxI  = uxR + NSEQ;

  qrnn_front<<<BB * TT, DD, 0, stream>>>(x, amp_w, amp_b, phase_w, phase_b, seqR, seqI);
  for (int layer = 0; layer < 2; ++layer) {          // shared cell params
    qrnn_uconjx<<<TT * BB, 256, 0, stream>>>(Ux, seqR, seqI, uxR, uxI);
    qrnn_scan<<<BB, 256, 0, stream>>>(Uh, lam_p, uxR, uxI, seqR, seqI);
  }
  qrnn_tail<<<TT * BB, 128, 0, stream>>>(seqR, seqI, dense_w, dlam_p, meas, (float*)d_out);
}
